// MultiScaleTransformerGSM_57655640981776
// MI455X (gfx1250) — matrix operand order repaired and checked by reading, not compiled, not run on hardware
//
#include <hip/hip_runtime.h>
#include <hip/hip_bf16.h>

typedef __attribute__((ext_vector_type(16))) __bf16 v16bf;
typedef __attribute__((ext_vector_type(8)))  float  v8f;
typedef unsigned short u16;

#define C_ 768
#define N_ 197
#define T_ 16
#define B_ 8
#define M_ (B_*T_*N_)   /* 25216 = 197*128 */
#define P_ 256
#define GK C_
#define KTILE 64
#define KPAD 72         /* 144B row stride: 16B-aligned chunks, spread banks */

__device__ __forceinline__ u16 f2bf(float f) {
  unsigned u = __float_as_uint(f);
  u += 0x7FFFu + ((u >> 16) & 1u);      // round-to-nearest-even
  return (u16)(u >> 16);
}
__device__ __forceinline__ float bf2f(u16 h) {
  return __uint_as_float(((unsigned)h) << 16);
}

// ---- weight transpose + bf16 pack: wT[n][k] = bf16(w[k][n]) ----------------
__global__ void wtrans_kernel(const float* __restrict__ w, u16* __restrict__ wT) {
  int idx = blockIdx.x * 256 + threadIdx.x;       // idx = n*768 + k
  int n = idx / C_, k = idx % C_;
  wT[idx] = f2bf(w[(size_t)k * C_ + n]);
}

// ---- LayerNorm + ReLU, one block per row of 768 ----------------------------
__global__ void ln_relu_kernel(const float* __restrict__ x, const float* __restrict__ g,
                               const float* __restrict__ b, float* __restrict__ xn) {
  int row = blockIdx.x, tid = threadIdx.x;
  const float* xr = x + (size_t)row * C_;
  float v0 = xr[tid], v1 = xr[tid + 256], v2 = xr[tid + 512];
  float s = v0 + v1 + v2, q = v0*v0 + v1*v1 + v2*v2;
  __shared__ float rs[256], rq[256];
  rs[tid] = s; rq[tid] = q; __syncthreads();
  for (int st = 128; st > 0; st >>= 1) {
    if (tid < st) { rs[tid] += rs[tid + st]; rq[tid] += rq[tid + st]; }
    __syncthreads();
  }
  float mean = rs[0] * (1.f / C_);
  float var  = rq[0] * (1.f / C_) - mean * mean;
  float inv  = rsqrtf(var + 1e-5f);
  float* o = xn + (size_t)row * C_;
  float t;
  t = (v0-mean)*inv*g[tid]     + b[tid];     o[tid]     = fmaxf(t, 0.f);
  t = (v1-mean)*inv*g[tid+256] + b[tid+256]; o[tid+256] = fmaxf(t, 0.f);
  t = (v2-mean)*inv*g[tid+512] + b[tid+512]; o[tid+512] = fmaxf(t, 0.f);
}

// ---- depthwise temporal conv per branch, output packed bf16 ----------------
__global__ void conv_pack_kernel(const float* __restrict__ xn,
    const float* __restrict__ w0, const float* __restrict__ b0,
    const float* __restrict__ w1, const float* __restrict__ b1,
    const float* __restrict__ w2, const float* __restrict__ b2,
    u16* __restrict__ f0, u16* __restrict__ f1, u16* __restrict__ f2o) {
  long long idx = (long long)blockIdx.x * 256 + threadIdx.x;
  int c = (int)(idx % C_); long long r = idx / C_;
  int n = (int)(r % N_); r /= N_;
  int t = (int)(r % T_); r /= T_;
  int b = (int)(r % B_); int br = (int)(r / B_);
  const float* cw; const float* cb; u16* out; int d = br + 1;
  if      (br == 0) { cw = w0; cb = b0; out = f0;  }
  else if (br == 1) { cw = w1; cb = b1; out = f1;  }
  else              { cw = w2; cb = b2; out = f2o; }
  int K = 2*d + 1;
  float acc = cb[c];
  for (int j = 0; j < K; ++j) {
    int tt = t + j - d;                 // lax correlation, pad (d,d)
    if (tt >= 0 && tt < T_)
      acc += xn[((size_t)(b*T_ + tt)*N_ + n)*C_ + c] * cw[c*K + j];
  }
  out[((size_t)(b*T_ + t)*N_ + n)*C_ + c] = f2bf(acc);
}

// ---- temporal shift gather + softmax(fusion_w) weighted sum ----------------
__global__ void fuse_kernel(const u16* __restrict__ y0, const u16* __restrict__ y1,
                            const u16* __restrict__ y2, const float* __restrict__ fw,
                            u16* __restrict__ fused) {
  long long idx = (long long)blockIdx.x * 256 + threadIdx.x;
  int c = (int)(idx % C_); long long r = idx / C_;
  int n = (int)(r % N_); r /= N_;
  int t = (int)(r % T_); int b = (int)(r / T_);
  float f0 = fw[0], f1 = fw[1], f2 = fw[2];
  float mx = fmaxf(f0, fmaxf(f1, f2));
  float e0 = __expf(f0-mx), e1 = __expf(f1-mx), e2 = __expf(f2-mx);
  float inv = 1.f / (e0 + e1 + e2);
  float w[3] = { e0*inv, e1*inv, e2*inv };
  const u16* ys[3] = { y0, y1, y2 };
  float acc = 0.f;
  #pragma unroll
  for (int i = 0; i < 3; ++i) {
    int d = i + 1;
    int ts = (c < P_) ? (t + d) : ((c < 2*P_) ? (t - d) : t);
    if (ts >= 0 && ts < T_)
      acc += w[i] * bf2f(ys[i][((size_t)(b*T_ + ts)*N_ + n)*C_ + c]);
  }
  fused[idx] = f2bf(acc);
}

// ---- bf16 WMMA GEMM: D = A(MxK) * Bw^T(NxK) + epilogue ---------------------
// Block tile 128x128, 8 waves (4M x 2N), wave tile 32x64, K step 64.
// EPI 0: y = sigmoid(acc+bias)*X  -> bf16    (gate branch)
// EPI 1: y = gelu_exact(acc+bias) -> bf16    (proj1)
// EPI 2: y = X + acc + bias       -> f32     (proj2 + residual)
template <int EPI>
__global__ __launch_bounds__(256) void gemm768_kernel(
    const u16* __restrict__ A, const u16* __restrict__ Bw,
    const float* __restrict__ bias, const float* __restrict__ X,
    void* __restrict__ Out)
{
  __shared__ u16 sA[128 * KPAD];   // [m][k], 18 KB
  __shared__ u16 sB[128 * KPAD];   // [n][k], 18 KB
  int tid  = threadIdx.x;
  int lane = tid & 31, wv = tid >> 5;
  int wm = wv & 3, wn = wv >> 2;         // 4x2 wave grid over 128x128 tile
  int l15 = lane & 15, hl = lane >> 4;   // half-wave select
  int mBase = blockIdx.x * 128;
  int nBase = blockIdx.y * 128;

  v8f acc[2][4];
  #pragma unroll
  for (int mi = 0; mi < 2; ++mi)
    #pragma unroll
    for (int ni = 0; ni < 4; ++ni)
      #pragma unroll
      for (int r = 0; r < 8; ++r) acc[mi][ni][r] = 0.f;

  for (int k0 = 0; k0 < GK; k0 += KTILE) {
    __syncthreads();
    // stage A & B tiles: 128 rows x 64 k each = 1024 16B-chunks per tile
    #pragma unroll
    for (int j = 0; j < 4; ++j) {
      int ch = tid + j * 256;
      int r = ch >> 3, kc = (ch & 7) * 8;
      *(uint4*)&sA[r * KPAD + kc] =
          *(const uint4*)&A[(size_t)(mBase + r) * GK + k0 + kc];
      *(uint4*)&sB[r * KPAD + kc] =
          *(const uint4*)&Bw[(size_t)(nBase + r) * GK + k0 + kc];
    }
    __syncthreads();

    union FragU { uint4 q[2]; v16bf v; };
    #pragma unroll
    for (int kk = 0; kk < KTILE; kk += 32) {
      FragU a[2], bfr[4];
      #pragma unroll
      for (int mi = 0; mi < 2; ++mi) {   // A frag: K {0..7,16..23}/{8..15,24..31}
        const u16* pr = &sA[(wm*32 + mi*16 + l15) * KPAD + kk + hl * 8];
        a[mi].q[0] = *(const uint4*)(pr);
        a[mi].q[1] = *(const uint4*)(pr + 16);
      }
      #pragma unroll
      for (int ni = 0; ni < 4; ++ni) {   // B frag: col n, K {0..7,16..23}/{8..15,24..31}
        const u16* pc = &sB[(wn*64 + ni*16 + l15) * KPAD + kk + hl * 8];
        bfr[ni].q[0] = *(const uint4*)(pc);
        bfr[ni].q[1] = *(const uint4*)(pc + 16);
      }
      #pragma unroll
      for (int mi = 0; mi < 2; ++mi)
        #pragma unroll
        for (int ni = 0; ni < 4; ++ni)
          acc[mi][ni] = __builtin_amdgcn_wmma_f32_16x16x32_bf16(
              false, a[mi].v, false, bfr[ni].v, (short)0, acc[mi][ni],
              false, false);
    }
  }

  #pragma unroll
  for (int mi = 0; mi < 2; ++mi)
    #pragma unroll
    for (int ni = 0; ni < 4; ++ni) {
      int n = nBase + wn*64 + ni*16 + l15;
      #pragma unroll
      for (int r = 0; r < 8; ++r) {
        int m = mBase + wm*32 + mi*16 + hl*8 + r;   // D layout: vgpr r, half hl
        float v = acc[mi][ni][r];
        size_t o = (size_t)m * GK + n;
        if (EPI == 0) {
          float gs = 1.f / (1.f + __expf(-(v + bias[n])));
          ((u16*)Out)[o] = f2bf(gs * X[o]);
        } else if (EPI == 1) {
          float tg = v + bias[n];
          ((u16*)Out)[o] = f2bf(0.5f * tg * (1.f + erff(tg * 0.70710678118f)));
        } else {
          ((float*)Out)[o] = X[o] + v + bias[n];
        }
      }
    }
}

extern "C" void kernel_launch(void* const* d_in, const int* in_sizes, int n_in,
                              void* d_out, int out_size, void* d_ws, size_t ws_size,
                              hipStream_t stream) {
  (void)in_sizes; (void)n_in; (void)out_size; (void)ws_size;
  const float* x      = (const float*)d_in[0];
  const float* ln_g   = (const float*)d_in[1];
  const float* ln_b   = (const float*)d_in[2];
  const float* gate_w = (const float*)d_in[3];
  const float* gate_b = (const float*)d_in[4];
  const float* pw1    = (const float*)d_in[5];
  const float* pb1    = (const float*)d_in[6];
  const float* pw2    = (const float*)d_in[7];
  const float* pb2    = (const float*)d_in[8];
  const float* fw     = (const float*)d_in[9];
  // d_in[10] = num_frames (hardcoded T_=16)
  const float* cw0 = (const float*)d_in[11];
  const float* cb0 = (const float*)d_in[12];
  const float* cw1 = (const float*)d_in[13];
  const float* cb1 = (const float*)d_in[14];
  const float* cw2 = (const float*)d_in[15];
  const float* cb2 = (const float*)d_in[16];

  const size_t MC = (size_t)M_ * C_;
  char* wsb = (char*)d_ws;
  size_t off = 0;
  auto carve = [&](size_t bytes) {
    char* p = wsb + off;
    off += (bytes + 255) & ~(size_t)255;
    return p;
  };
  float* xn  = (float*)carve(MC * 4);
  u16* feat0 = (u16*)carve(MC * 2);
  u16* feat1 = (u16*)carve(MC * 2);
  u16* feat2 = (u16*)carve(MC * 2);
  u16* y0    = (u16*)carve(MC * 2);
  u16* y1    = (u16*)carve(MC * 2);
  u16* y2    = (u16*)carve(MC * 2);
  u16* fused = (u16*)carve(MC * 2);
  u16* hB    = (u16*)carve(MC * 2);
  u16* wTg   = (u16*)carve((size_t)C_ * C_ * 2);
  u16* wT1   = (u16*)carve((size_t)C_ * C_ * 2);
  u16* wT2   = (u16*)carve((size_t)C_ * C_ * 2);

  wtrans_kernel<<<(C_ * C_) / 256, 256, 0, stream>>>(gate_w, wTg);
  wtrans_kernel<<<(C_ * C_) / 256, 256, 0, stream>>>(pw1, wT1);
  wtrans_kernel<<<(C_ * C_) / 256, 256, 0, stream>>>(pw2, wT2);

  ln_relu_kernel<<<M_, 256, 0, stream>>>(x, ln_g, ln_b, xn);

  conv_pack_kernel<<<(unsigned)((3 * MC) / 256), 256, 0, stream>>>(
      xn, cw0, cb0, cw1, cb1, cw2, cb2, feat0, feat1, feat2);

  dim3 ggrid(M_ / 128, C_ / 128);   // 197 x 6
  gemm768_kernel<0><<<ggrid, 256, 0, stream>>>(feat0, wTg, gate_b, x, y0);
  gemm768_kernel<0><<<ggrid, 256, 0, stream>>>(feat1, wTg, gate_b, x, y1);
  gemm768_kernel<0><<<ggrid, 256, 0, stream>>>(feat2, wTg, gate_b, x, y2);

  fuse_kernel<<<(unsigned)(MC / 256), 256, 0, stream>>>(y0, y1, y2, fw, fused);

  gemm768_kernel<1><<<ggrid, 256, 0, stream>>>(fused, wT1, pb1, nullptr, hB);
  gemm768_kernel<2><<<ggrid, 256, 0, stream>>>(hB, wT2, pb2, x, d_out);
}
